// AttMix_833223655697
// MI455X (gfx1250) — matrix operand order repaired and checked by reading, not compiled, not run on hardware
//
#include <hip/hip_runtime.h>
#include <hip/hip_bf16.h>
#include <stdint.h>

#define B_     1024
#define L_     200
#define H_     128
#define HEADS_ 8
#define NNODE_ 200000
#define MAXN_  201
#define EPS_   1e-12f

typedef __attribute__((ext_vector_type(16))) _Float16 v16h;
typedef __attribute__((ext_vector_type(8)))  float    v8f;
typedef __attribute__((ext_vector_type(4)))  _Float16 h4;

__device__ __forceinline__ float waveRedSum(float x) {
#pragma unroll
    for (int m = 16; m > 0; m >>= 1) x += __shfl_xor(x, m, 32);
    return x;
}
__device__ __forceinline__ int waveRedSumI(int x) {
#pragma unroll
    for (int m = 16; m > 0; m >>= 1) x += __shfl_xor(x, m, 32);
    return x;
}

// ---------------------------------------------------------------------------
// Kernel 1: normalize emb_W rows and write transposed b.T (128 x 199999).
// LDS-tiled: 64 rows x 128 cols staged; reads float4-coalesced, writes
// 64-wide contiguous runs per h (256B coalesced).
// ---------------------------------------------------------------------------
__global__ __launch_bounds__(256) void k_norm_emb(const float* __restrict__ emb,
                                                  float* __restrict__ bT) {
    __shared__ float tile[64][129];  // +1 pad kills bank conflicts on transpose read
    const int n0 = blockIdx.x * 64;
    const int t = threadIdx.x, wid = t >> 5, lane = t & 31;
#pragma unroll
    for (int r8 = 0; r8 < 8; ++r8) {
        const int rl = wid * 8 + r8;
        const int row = n0 + rl;                   // always < NNODE_ (3125*64 exact)
        const float4 e = ((const float4*)(emb + (size_t)row * H_))[lane];
        float ss = e.x * e.x + e.y * e.y + e.z * e.z + e.w * e.w;
        ss = waveRedSum(ss);
        const float inv = 1.0f / (sqrtf(ss) + EPS_);
        tile[rl][lane * 4 + 0] = e.x * inv;
        tile[rl][lane * 4 + 1] = e.y * inv;
        tile[rl][lane * 4 + 2] = e.z * inv;
        tile[rl][lane * 4 + 3] = e.w * inv;
    }
    __syncthreads();
#pragma unroll
    for (int i = 0; i < 32; ++i) {
        const int idx = t + 256 * i;               // 0..8191
        const int h = idx >> 6, nl = idx & 63;
        const int row = n0 + nl;
        if (row >= 1) bT[(size_t)h * (NNODE_ - 1) + (row - 1)] = tile[nl][h];
    }
}

// ---------------------------------------------------------------------------
// Kernel 2: seq gather + double-normalize + mask -> f16 (B*L, 128).
// One wave per (b,l) row; masked rows store exact zeros.
// ---------------------------------------------------------------------------
__global__ __launch_bounds__(256) void k_seq(const int* __restrict__ alias,
                                             const int* __restrict__ items,
                                             const int* __restrict__ mask,
                                             const float* __restrict__ emb,
                                             _Float16* __restrict__ seqh) {
    const int t = threadIdx.x, wid = t >> 5, lane = t & 31;
    const int row = blockIdx.x * 8 + wid;          // 0..B*L-1
    const int b = row / L_, l = row - b * L_;
    h4 outv;
    outv.x = (_Float16)0.0f; outv.y = (_Float16)0.0f;
    outv.z = (_Float16)0.0f; outv.w = (_Float16)0.0f;
    if (mask[b * L_ + l]) {
        const int al = alias[b * L_ + l];
        const int item = items[b * MAXN_ + al];
        const float4 e = ((const float4*)(emb + (size_t)item * H_))[lane];
        float ss = waveRedSum(e.x * e.x + e.y * e.y + e.z * e.z + e.w * e.w);
        const float inv1 = 1.0f / (sqrtf(ss) + EPS_);
        float x = e.x * inv1, y = e.y * inv1, z = e.z * inv1, w = e.w * inv1;
        float ss2 = waveRedSum(x * x + y * y + z * z + w * w);
        const float inv2 = 1.0f / (sqrtf(ss2) + EPS_);
        x *= inv2; y *= inv2; z *= inv2; w *= inv2;
        outv.x = (_Float16)x; outv.y = (_Float16)y;
        outv.z = (_Float16)z; outv.w = (_Float16)w;
    }
    ((h4*)seqh)[(size_t)row * 32 + lane] = outv;
}

// ---------------------------------------------------------------------------
// Kernel 3: ht_last (B,128) f32 — recompute last row per b (avoids f32 seq buf).
// ---------------------------------------------------------------------------
__global__ __launch_bounds__(256) void k_htlast(const int* __restrict__ alias,
                                                const int* __restrict__ items,
                                                const int* __restrict__ mask,
                                                const float* __restrict__ emb,
                                                float* __restrict__ ht) {
    const int t = threadIdx.x, wid = t >> 5, lane = t & 31;
    const int b = blockIdx.x * 8 + wid;
    int cnt = 0;
    for (int i = lane; i < L_; i += 32) cnt += mask[b * L_ + i];
    cnt = waveRedSumI(cnt);
    const int last = cnt - 1;                      // lengths >= 1 guaranteed
    const int al = alias[b * L_ + last];
    const int item = items[b * MAXN_ + al];
    const float4 e = ((const float4*)(emb + (size_t)item * H_))[lane];
    float ss = waveRedSum(e.x * e.x + e.y * e.y + e.z * e.z + e.w * e.w);
    const float inv1 = 1.0f / (sqrtf(ss) + EPS_);
    float x = e.x * inv1, y = e.y * inv1, z = e.z * inv1, w = e.w * inv1;
    float ss2 = waveRedSum(x * x + y * y + z * z + w * w);
    const float inv2 = 1.0f / (sqrtf(ss2) + EPS_);
    float4 o; o.x = x * inv2; o.y = y * inv2; o.z = z * inv2; o.w = w * inv2;
    ((float4*)ht)[(size_t)b * 32 + lane] = o;
}

// ---------------------------------------------------------------------------
// Kernel 4: hts = elemwise-sign-norm(ht @ Wq^T + bq); q0 = hts @ W0^T + b0.
// ---------------------------------------------------------------------------
__global__ __launch_bounds__(128) void k_q0(const float* __restrict__ ht,
                                            const float* __restrict__ Wq,
                                            const float* __restrict__ bq,
                                            const float* __restrict__ W0,
                                            const float* __restrict__ b0,
                                            float* __restrict__ q0) {
    __shared__ float hs[128], ssh[128];
    const int b = blockIdx.x, t = threadIdx.x;
    hs[t] = ht[(size_t)b * H_ + t];
    __syncthreads();
    float acc = bq[t];
    const float* w = Wq + (size_t)t * H_;
#pragma unroll 4
    for (int k = 0; k < H_; ++k) acc += w[k] * hs[k];
    ssh[t] = acc / (fabsf(acc) + EPS_);            // norm over size-1 axis == |x|
    __syncthreads();
    float a2 = b0[t];
    const float* w0 = W0 + (size_t)t * H_;
#pragma unroll 4
    for (int k = 0; k < H_; ++k) a2 += w0[k] * ssh[k];
    q0[(size_t)b * H_ + t] = a2;
}

// ---------------------------------------------------------------------------
// Kernel 5: pack W (H x H, f32, row = output n) into WMMA B-fragment order
// (f16). Consumed as 8 contiguous dwords per lane per (ntile, kstep).
// B order packed here (contiguous k, same as A): lanes 0-15 K=0..15, lanes 16-31 K=16..31,
// VGPR v holds K=2v (lo), 2v+1 (hi); N = lane&15.
// ---------------------------------------------------------------------------
__global__ __launch_bounds__(256) void k_pack(const float* __restrict__ W,
                                              unsigned int* __restrict__ Wpk) {
    const int t = threadIdx.x;
#pragma unroll
    for (int i = 0; i < 32; ++i) {
        const int flat = t + 256 * i;              // 0..8191
        const int v = flat & 7;
        const int lane = (flat >> 3) & 31;
        const int ks = (flat >> 8) & 3;
        const int nt = (flat >> 10) & 7;
        const int n = nt * 16 + (lane & 15);
        const int k0 = ks * 32 + ((lane < 16) ? 0 : 16) + 2 * v;
        union { _Float16 h[2]; unsigned int u; } c;
        c.h[0] = (_Float16)W[n * H_ + k0];
        c.h[1] = (_Float16)W[n * H_ + k0 + 1];
        Wpk[flat] = c.u;
    }
}

// ---------------------------------------------------------------------------
// Kernel 6: fused dual GEMM  q1 = seq@W1^T + b1, q2 = seq@W2^T + b2 via
// v_wmma_f32_16x16x32_f16. A (16x128 f16 panel) staged in LDS once, each wave
// owns one 16-wide N-tile and issues TWO WMMAs per k-step (shared A fragment)
// -> seq_h is streamed from HBM once instead of twice (-52 MB traffic).
// Epilogue restages D tiles in LDS so f16 outputs leave as 256B-coalesced
// global_store_b128 instead of scattered b16 stores.
// ---------------------------------------------------------------------------
__global__ __launch_bounds__(256) void k_gemm(const _Float16* __restrict__ Ah,
                                              const unsigned int* __restrict__ Bpk1,
                                              const unsigned int* __restrict__ Bpk2,
                                              const float* __restrict__ bias1,
                                              const float* __restrict__ bias2,
                                              _Float16* __restrict__ O1,
                                              _Float16* __restrict__ O2) {
    __shared__ _Float16 As[16 * 128];              // 4KB: A panel, then D tiles
    const int t = threadIdx.x;
    const size_t m0 = (size_t)blockIdx.x * 16;
    ((uint4*)As)[t] = ((const uint4*)(Ah + m0 * H_))[t];
    __syncthreads();

    const int w = t >> 5, lane = t & 31;
    const int M = lane & 15;
    const unsigned int* As32 = (const unsigned int*)As;
    v8f acc1 = {}, acc2 = {};
#pragma unroll
    for (int ks = 0; ks < 4; ++ks) {
        // A frag (16-bit 16x32): lanes 0-15 K=0..15; lanes 16-31 K=16..31;
        // (2v,2v+1) pairs contiguous -> b32 LDS reads (compiler fuses to b128)
        union { unsigned int u[8]; v16h v; } af;
        const int kb = ks * 32 + ((lane < 16) ? 0 : 16);
#pragma unroll
        for (int vv = 0; vv < 8; ++vv) {
            const int off = 2 * vv;
            af.u[vv] = As32[M * 64 + ((kb + off) >> 1)];
        }
        const int bidx = ((w * 4 + ks) * 32 + lane) << 3;
        union { uint4 q[2]; v16h v; } bf1, bf2;
        const uint4* bp1 = (const uint4*)(Bpk1 + bidx);
        bf1.q[0] = bp1[0]; bf1.q[1] = bp1[1];
        const uint4* bp2 = (const uint4*)(Bpk2 + bidx);
        bf2.q[0] = bp2[0]; bf2.q[1] = bp2[1];
        acc1 = __builtin_amdgcn_wmma_f32_16x16x32_f16(
            false, af.v, false, bf1.v, (short)0, acc1, false, false);
        acc2 = __builtin_amdgcn_wmma_f32_16x16x32_f16(
            false, af.v, false, bf2.v, (short)0, acc2, false, false);
    }
    // D layout: VGPR r -> lanes 0-15: M=r, lanes 16-31: M=8+r; N = w*16+lane%16
    const int N = w * 16 + (lane & 15);
    const int mb = (lane < 16) ? 0 : 8;
    const float bs1 = bias1[N], bs2 = bias2[N];

    __syncthreads();                               // A-frag reads done; reuse As
#pragma unroll
    for (int r = 0; r < 8; ++r)
        As[(mb + r) * H_ + N] = (_Float16)(acc1[r] + bs1);
    __syncthreads();
    ((uint4*)(O1 + m0 * H_))[t] = ((const uint4*)As)[t];
    __syncthreads();
#pragma unroll
    for (int r = 0; r < 8; ++r)
        As[(mb + r) * H_ + N] = (_Float16)(acc2[r] + bs2);
    __syncthreads();
    ((uint4*)(O2 + m0 * H_))[t] = ((const uint4*)As)[t];
}

// ---------------------------------------------------------------------------
// Kernel 7: attention + LayerNorm + Wt + L2 norm. One block per b.
// Reproduces the raw C-order q1.reshape(B*8, L, 16):
//   alpha[b,j,l] = sum_d q0[b,j*16+d] * q1[b, j*25 + (l>>3), 16*(l&7)+d]
// ---------------------------------------------------------------------------
__global__ __launch_bounds__(256) void k_attn(const _Float16* __restrict__ q1h,
                                              const _Float16* __restrict__ q2h,
                                              const float* __restrict__ q0,
                                              const float* __restrict__ ht,
                                              const int* __restrict__ mask,
                                              const float* __restrict__ lng,
                                              const float* __restrict__ lnb,
                                              const float* __restrict__ Wt,
                                              const float* __restrict__ bt,
                                              float* __restrict__ outA) {
    __shared__ float q0s[128], htl[128], alphaS[L_ * 8], mk[L_];
    __shared__ float sred[64], denom[8], av[128], aln[128], r1[256], r2[256];
    const int b = blockIdx.x, t = threadIdx.x;

    if (t < 128) {
        q0s[t] = q0[(size_t)b * H_ + t];
        htl[t] = ht[(size_t)b * H_ + t];
    }
    for (int i = t; i < L_; i += 256) mk[i] = (float)mask[b * L_ + i];
    __syncthreads();

    // exp(2*sigmoid(q0 . q1)) for all (l, head j) — softmax over ALL l
    for (int idx = t; idx < L_ * 8; idx += 256) {
        const int l = idx >> 3, j = idx & 7;
        const int lrow = j * 25 + (l >> 3);
        const int cb = (l & 7) * 16;
        const _Float16* p = q1h + ((size_t)(b * L_ + lrow)) * H_ + cb;
        float s = 0.0f;
#pragma unroll
        for (int d = 0; d < 16; ++d) s += (float)p[d] * q0s[j * 16 + d];
        const float sg = 1.0f / (1.0f + expf(-s));
        alphaS[l * 8 + j] = expf(2.0f * sg);
    }
    __syncthreads();

    // softmax denominators per head
    if (t < 64) {
        const int sub = t >> 3, j = t & 7;
        float s = 0.0f;
        for (int l = sub; l < L_; l += 8) s += alphaS[l * 8 + j];
        sred[t] = s;
    }
    __syncthreads();
    if (t < 8) {
        float s = 0.0f;
#pragma unroll
        for (int g = 0; g < 8; ++g) s += sred[g * 8 + t];
        denom[t] = s;
    }
    __syncthreads();

    // masked weighted sum over l (coalesced 256B q2 rows per iteration)
    if (t < 128) {
        const int j = t >> 4;
        float accv = 0.0f;
        for (int l = 0; l < L_; ++l)
            accv += mk[l] * alphaS[l * 8 + j] *
                    (float)q2h[((size_t)(b * L_ + l)) * H_ + t];
        av[t] = accv / denom[j];
    }
    __syncthreads();

    // LayerNorm over 128
    const float x = (t < 128) ? av[t] : 0.0f;
    r1[t] = x; r2[t] = x * x;
    __syncthreads();
    for (int s2 = 128; s2 > 0; s2 >>= 1) {
        if (t < s2) { r1[t] += r1[t + s2]; r2[t] += r2[t + s2]; }
        __syncthreads();
    }
    const float mean = r1[0] * (1.0f / 128.0f);
    const float var = r2[0] * (1.0f / 128.0f) - mean * mean;
    const float rstd = rsqrtf(var + 1e-8f);
    if (t < 128) aln[t] = (av[t] - mean) * rstd * lng[t] + lnb[t];
    __syncthreads();

    // out = L2norm( [aln, ht] @ Wt^T + bt )
    float acc = 0.0f;
    if (t < 128) {
        acc = bt[t];
        const float* w = Wt + (size_t)t * (2 * H_);
#pragma unroll 4
        for (int k = 0; k < H_; ++k) acc += w[k] * aln[k];
#pragma unroll 4
        for (int k = 0; k < H_; ++k) acc += w[H_ + k] * htl[k];
    }
    r2[t] = acc * acc;
    __syncthreads();
    for (int s2 = 128; s2 > 0; s2 >>= 1) {
        if (t < s2) r2[t] += r2[t + s2];
        __syncthreads();
    }
    const float nrm = sqrtf(r2[0]) + EPS_;
    if (t < 128) outA[(size_t)b * H_ + t] = acc / nrm;
}

// ---------------------------------------------------------------------------
extern "C" void kernel_launch(void* const* d_in, const int* in_sizes, int n_in,
                              void* d_out, int out_size, void* d_ws, size_t ws_size,
                              hipStream_t stream) {
    (void)in_sizes; (void)n_in; (void)out_size; (void)ws_size;
    const int*   alias = (const int*)d_in[0];
    const int*   items = (const int*)d_in[1];
    const int*   mask  = (const int*)d_in[2];
    const float* emb   = (const float*)d_in[3];
    const float* W0    = (const float*)d_in[4];
    const float* b0    = (const float*)d_in[5];
    const float* W1    = (const float*)d_in[6];
    const float* b1    = (const float*)d_in[7];
    const float* W2    = (const float*)d_in[8];
    const float* b2    = (const float*)d_in[9];
    const float* Wq    = (const float*)d_in[10];
    const float* bq    = (const float*)d_in[11];
    const float* Wt    = (const float*)d_in[12];
    const float* bt    = (const float*)d_in[13];
    const float* lng   = (const float*)d_in[14];
    const float* lnb   = (const float*)d_in[15];

    float* outA  = (float*)d_out;
    float* outBT = outA + (size_t)B_ * H_;

    char* ws = (char*)d_ws;
    size_t off = 0;
    const size_t seqBytes = (size_t)B_ * L_ * H_ * sizeof(_Float16);
    _Float16* seqh = (_Float16*)(ws + off); off += seqBytes;
    _Float16* q1h  = (_Float16*)(ws + off); off += seqBytes;
    _Float16* q2h  = (_Float16*)(ws + off); off += seqBytes;
    float* htl = (float*)(ws + off); off += (size_t)B_ * H_ * sizeof(float);
    float* q0v = (float*)(ws + off); off += (size_t)B_ * H_ * sizeof(float);
    unsigned int* W1pk = (unsigned int*)(ws + off); off += 8192 * sizeof(unsigned int);
    unsigned int* W2pk = (unsigned int*)(ws + off); off += 8192 * sizeof(unsigned int);

    k_pack<<<1, 256, 0, stream>>>(W1, W1pk);
    k_pack<<<1, 256, 0, stream>>>(W2, W2pk);
    k_norm_emb<<<NNODE_ / 64, 256, 0, stream>>>(emb, outBT);
    k_seq<<<(B_ * L_) / 8, 256, 0, stream>>>(alias, items, mask, emb, seqh);
    k_htlast<<<B_ / 8, 256, 0, stream>>>(alias, items, mask, emb, htl);
    k_q0<<<B_, 128, 0, stream>>>(htl, Wq, bq, W0, b0, q0v);
    k_gemm<<<(B_ * L_) / 16, 256, 0, stream>>>(seqh, W1pk, W2pk, b1, b2, q1h, q2h);
    k_attn<<<B_, 256, 0, stream>>>(q1h, q2h, q0v, htl, mask, lng, lnb, Wt, bt, outA);
}
